// My_Loss_60327110639930
// MI455X (gfx1250) — matrix operand order repaired and checked by reading, not compiled, not run on hardware
//
#include <hip/hip_runtime.h>
#include <math.h>

#define NCLS 101

typedef __attribute__((ext_vector_type(16))) _Float16 v16h;
typedef __attribute__((ext_vector_type(8)))  float    v8f;

// ---------------------------------------------------------------------------
// Workspace layout (W = (float*)d_ws):
//   W[0]  clsSum (float, atomic)
//   W[1]  sumPos (float, atomic)
//   W[2]  sumNeg (float, atomic)
//   I[3]  posCnt (int, atomic)            I = (int*)d_ws
//   I[4]  M      (int, #correct rows, atomic counter)
//   W[16 .. 16+B)        invn[g]  : 1/||hash_feature[list[g]]||
//   I[16+B .. 16+2B)     list[g]  : original row index of g-th correct row
// ---------------------------------------------------------------------------

// ---- WMMA operand builders (f32 -> f16 on the fly) ------------------------
// A (16x32 f16): lane r=lane&15, sel=lane>>4 holds row m0+r,
//   elements 0..7  = K[k0+16*sel    .. +8)
//   elements 8..15 = K[k0+16*sel+8  .. +8)
__device__ __forceinline__ v16h make_a(const float* __restrict__ row, int k0, int sel) {
  v16h r;
  const float* p0 = row + k0 + sel * 16;
  const float* p1 = row + k0 + 8 + sel * 16;
#pragma unroll
  for (int q = 0; q < 8; ++q) { r[q] = (_Float16)p0[q]; r[8 + q] = (_Float16)p1[q]; }
  return r;
}
// B (32x16 f16): lane holds row n0+r, elements 0..15 = K[k0+16*sel .. +16)
__device__ __forceinline__ v16h make_b(const float* __restrict__ row, int k0, int sel) {
  v16h r;
  const float* p = row + k0 + sel * 16;
#pragma unroll
  for (int q = 0; q < 16; ++q) r[q] = (_Float16)p[q];
  return r;
}

__device__ __forceinline__ v8f wmma16(v16h a, v16h b, v8f c) {
  return __builtin_amdgcn_wmma_f32_16x16x32_f16(false, a, false, b,
                                                (short)0, c, false, false);
}

// ---- K0: init outputs + workspace accumulators ----------------------------
__global__ void init_kernel(const float* __restrict__ simNow,
                            const float* __restrict__ cntMat,
                            float* __restrict__ out, float* __restrict__ W) {
  int i = blockIdx.x * blockDim.x + threadIdx.x;
  if (i < NCLS * NCLS) {
    out[i] = simNow[i];
    out[NCLS * NCLS + i] = cntMat[i];
  }
  if (i < 3)  out[2 * NCLS * NCLS + i] = 0.0f;
  if (i < 16) W[i] = 0.0f;   // zeroes float and int counters (bit pattern 0)
}

// ---- K1: per-row log-softmax CE + argmax + compaction of correct rows -----
__global__ void cls_kernel(const float* __restrict__ cls, const int* __restrict__ tgt,
                           float* __restrict__ W, int B) {
  int wid  = (blockIdx.x * blockDim.x + threadIdx.x) >> 5;  // one wave32 per row
  int lane = threadIdx.x & 31;
  if (wid >= B) return;
  const float* row = cls + (size_t)wid * NCLS;

  float vmax = -3.4e38f; int imax = 0x7fffffff;
  for (int j = lane; j < NCLS; j += 32) {
    float x = row[j];
    if (x > vmax || (x == vmax && j < imax)) { vmax = x; imax = j; }
  }
  for (int off = 16; off; off >>= 1) {
    float ov = __shfl_xor(vmax, off);
    int   oi = __shfl_xor(imax, off);
    if (ov > vmax || (ov == vmax && oi < imax)) { vmax = ov; imax = oi; }
  }
  float se = 0.0f;
  for (int j = lane; j < NCLS; j += 32) se += expf(row[j] - vmax);
  for (int off = 16; off; off >>= 1) se += __shfl_xor(se, off);

  if (lane == 0) {
    int t = tgt[wid];
    atomicAdd(W, vmax + logf(se) - row[t]);          // CE term
    if (imax == t) {                                  // correct -> compact
      int* I = (int*)W;
      int p = atomicAdd(I + 4, 1);
      int* list = (int*)(W + 16 + B);
      list[p] = wid;
    }
  }
}

// ---- K2: WMMA theta tiles + weighted pair-loss reduction ------------------
// theta = H H^T / 2, H = hash_out (B x DH).  Each wave computes a 32x32
// region as 2x2 WMMA tiles with A/B operand reuse.  Upper-triangular blocks
// only; symmetric contributions doubled.
__global__ void __launch_bounds__(32)
theta_kernel(const float* __restrict__ H, const int* __restrict__ tgt,
             const float* __restrict__ simLast, float* __restrict__ W, int DH) {
  int bx = blockIdx.x, by = blockIdx.y;
  if (bx < by) return;                 // uniform per block: EXEC stays all-1s
  int lane = threadIdx.x, r = lane & 15, sel = lane >> 4;

  const float* ar0 = H + (size_t)(by * 32 + r) * DH;
  const float* ar1 = ar0 + (size_t)16 * DH;
  const float* br0 = H + (size_t)(bx * 32 + r) * DH;
  const float* br1 = br0 + (size_t)16 * DH;

  v8f c00 = {}, c01 = {}, c10 = {}, c11 = {};
  for (int k0 = 0; k0 < DH; k0 += 32) {
    v16h a0 = make_a(ar0, k0, sel);
    v16h a1 = make_a(ar1, k0, sel);
    v16h b0 = make_b(br0, k0, sel);
    v16h b1 = make_b(br1, k0, sel);
    c00 = wmma16(a0, b0, c00);
    c01 = wmma16(a0, b1, c01);
    c10 = wmma16(a1, b0, c10);
    c11 = wmma16(a1, b1, c11);
  }

  float sumP = 0.0f, sumN = 0.0f; int pc = 0;

  auto proc = [&](const v8f& c, int m0, int n0, float fs, int ps) {
    int j  = n0 + r;                   // this lane's output column
    int tj = tgt[j];
#pragma unroll
    for (int v = 0; v < 8; ++v) {
      int   i  = m0 + sel * 8 + v;     // this VGPR's output row
      int   ti = tgt[i];
      float th = c[v] * 0.5f;
      bool  pos = (ti == tj);
      float s   = pos ? 1.0f : 0.0f;
      if (i < 6 && j < 6) { s = simLast[ti * NCLS + tj]; pos = (s == 1.0f); }
      if (pos) pc += ps;
      if (i != j) {
        // stable softplus: log1p(exp(th)) = max(th,0) + log1p(exp(-|th|))
        float pl = fs * (fmaxf(th, 0.0f) + log1pf(expf(-fabsf(th))) - s * th);
        if (pos) sumP += pl; else sumN += pl;
      }
    }
  };

  int m0 = by * 32, n0 = bx * 32;
  if (bx > by) {                       // strictly upper: all 4 subtiles x2
    proc(c00, m0,      n0,      2.0f, 2);
    proc(c01, m0,      n0 + 16, 2.0f, 2);
    proc(c10, m0 + 16, n0,      2.0f, 2);
    proc(c11, m0 + 16, n0 + 16, 2.0f, 2);
  } else {                             // diagonal block
    proc(c00, m0,      n0,      1.0f, 1);
    proc(c01, m0,      n0 + 16, 2.0f, 2);   // (1,0) is its mirror: skipped
    proc(c11, m0 + 16, n0 + 16, 1.0f, 1);
  }

  for (int off = 16; off; off >>= 1) {
    sumP += __shfl_xor(sumP, off);
    sumN += __shfl_xor(sumN, off);
    pc   += __shfl_xor(pc,   off);
  }
  if (lane == 0) {
    atomicAdd(W + 1, sumP);
    atomicAdd(W + 2, sumN);
    atomicAdd(((int*)W) + 3, pc);
  }
}

// ---- K3: inverse norms of the gathered (correct) rows ---------------------
__global__ void norm_kernel(const float* __restrict__ hf, float* __restrict__ W,
                            int B, int DF) {
  __shared__ float red[8];
  int g = blockIdx.x;
  const int* I = (const int*)W;
  if (g >= I[4]) return;               // uniform per block
  const int* list = (const int*)(W + 16 + B);
  const float4* row = (const float4*)(hf + (size_t)list[g] * DF);
  float s = 0.0f;
  for (int k = threadIdx.x; k < DF / 4; k += 256) {
    float4 x = row[k];
    s += x.x * x.x + x.y * x.y + x.z * x.z + x.w * x.w;
  }
  for (int off = 16; off; off >>= 1) s += __shfl_xor(s, off);
  if ((threadIdx.x & 31) == 0) red[threadIdx.x >> 5] = s;
  __syncthreads();
  if (threadIdx.x == 0) {
    float t = 0.0f;
    for (int w = 0; w < 8; ++w) t += red[w];
    (W + 16)[g] = rsqrtf(t);
  }
}

// ---- K4: pairwise cosine sims among correct rows, scatter by class --------
__global__ void pair_kernel(const float* __restrict__ hf, const int* __restrict__ tgt,
                            const float* __restrict__ W, float* __restrict__ out,
                            int B, int DF) {
  const int* I = (const int*)W;
  int M = I[4];
  int tm = blockIdx.y, tn = blockIdx.x;
  if (tn < tm) return;
  if (tm * 32 >= M) return;
  const int*   list = (const int*)(W + 16 + B);
  const float* invn = W + 16;
  float* outSim = out;
  float* outCnt = out + NCLS * NCLS;

  for (int p = threadIdx.x; p < 1024; p += 256) {
    int m = tm * 32 + (p >> 5);
    int n = tn * 32 + (p & 31);
    if (n >= M || m >= n) continue;    // each unordered pair exactly once
    int i = list[m], j = list[n];
    const float4* ri = (const float4*)(hf + (size_t)i * DF);
    const float4* rj = (const float4*)(hf + (size_t)j * DF);
    __builtin_prefetch(ri, 0, 0);      // global_prefetch_b8
    __builtin_prefetch(rj, 0, 0);
    float d = 0.0f;
    for (int k = 0; k < DF / 4; ++k) {
      float4 x = ri[k], y = rj[k];
      d = fmaf(x.x, y.x, d); d = fmaf(x.y, y.y, d);
      d = fmaf(x.z, y.z, d); d = fmaf(x.w, y.w, d);
    }
    float val = d * invn[m] * invn[n];
    int a = tgt[i], b = tgt[j];
    atomicAdd(&outSim[a * NCLS + b], val);
    atomicAdd(&outCnt[a * NCLS + b], 1.0f);
    if (a != b) {
      atomicAdd(&outSim[b * NCLS + a], val);
      atomicAdd(&outCnt[b * NCLS + a], 1.0f);
    }
  }
}

// ---- K5: finalize scalars --------------------------------------------------
__global__ void final_kernel(const float* __restrict__ W, float* __restrict__ out, int B) {
  if (threadIdx.x != 0 || blockIdx.x != 0) return;
  float Bf = (float)B;
  float clsSum = W[0], sumP = W[1], sumN = W[2];
  int posCnt = ((const int*)W)[3];
  float S1 = (float)posCnt - Bf;
  float S0 = Bf * Bf - (float)posCnt;
  if (S0 == 0.0f) S0 = 1.0f;
  if (S1 == 0.0f) S1 = 1.0f;
  float S = S0 + S1;
  float cnt = Bf * (Bf - 1.0f) * 0.5f;
  float hash_loss = (sumP * (S / S1) + sumN * (S / S0)) * 0.5f / cnt;
  float cls_loss  = clsSum / Bf;
  out[2 * NCLS * NCLS + 0] = hash_loss;
  out[2 * NCLS * NCLS + 1] = cls_loss;
  out[2 * NCLS * NCLS + 2] = 1.0f * cls_loss + 0.01f * hash_loss;  // GAMM, ALPH
}

// ---------------------------------------------------------------------------
extern "C" void kernel_launch(void* const* d_in, const int* in_sizes, int n_in,
                              void* d_out, int out_size, void* d_ws, size_t ws_size,
                              hipStream_t stream) {
  const float* hf     = (const float*)d_in[0];  // hash_feature (B, DF)
  const float* ho     = (const float*)d_in[1];  // hash_out     (B, DH)
  const float* cls    = (const float*)d_in[2];  // cls_out      (B, NCLS)
  const int*   tgt    = (const int*)  d_in[3];  // target       (B,)
  const float* simL   = (const float*)d_in[4];  // sim_matrix_last
  const float* simNow = (const float*)d_in[5];  // sim_matrix_now
  const float* cntMat = (const float*)d_in[6];  // count_matrix
  float* out = (float*)d_out;
  float* W   = (float*)d_ws;

  int B  = in_sizes[3];
  int DH = in_sizes[1] / B;   // 64
  int DF = in_sizes[0] / B;   // 2048

  init_kernel<<<(NCLS * NCLS + 255) / 256, 256, 0, stream>>>(simNow, cntMat, out, W);
  cls_kernel<<<(B * 32 + 255) / 256, 256, 0, stream>>>(cls, tgt, W, B);
  dim3 tg(B / 32, B / 32);
  theta_kernel<<<tg, 32, 0, stream>>>(ho, tgt, simL, W, DH);
  norm_kernel<<<B, 256, 0, stream>>>(hf, W, B, DF);
  dim3 pg((B + 31) / 32, (B + 31) / 32);
  pair_kernel<<<pg, 256, 0, stream>>>(hf, tgt, W, out, B, DF);
  final_kernel<<<1, 32, 0, stream>>>(W, out, B);
}
